// LinearConvAttention_27848567947730
// MI455X (gfx1250) — matrix operand order repaired and checked by reading, not compiled, not run on hardware
//
#include <hip/hip_runtime.h>
#include <hip/hip_bf16.h>

// ---------------------------------------------------------------------------
// Problem constants (from reference): x(16,192,128,128) f32
// ---------------------------------------------------------------------------
#define BATCH   16
#define NCH     192
#define HIDW    128
#define NHEADS  4
#define DKEYS   32
#define NPIX    16384        // 128*128
#define QKV3    384          // 3*HID
#define GN_EPS  1e-5f
#define SM_SCALE 0.17677669529663687f   // 32^-0.5

typedef __bf16 bf16_t;
typedef __attribute__((ext_vector_type(16))) __bf16 v16bf;
typedef __attribute__((ext_vector_type(8)))  __bf16 v8bf;
typedef __attribute__((ext_vector_type(8)))  float  v8f;
typedef __attribute__((ext_vector_type(4)))  float  v4f;
typedef __attribute__((ext_vector_type(4)))  int    v4i;

__device__ __forceinline__ unsigned short bf2us(bf16_t b) {
  return __builtin_bit_cast(unsigned short, b);
}

// 16-byte global->LDS copy. Uses the CDNA5 async DMA path when the toolchain
// exposes it (GLOBAL_LOAD_ASYNC_TO_LDS_B128, tracked by ASYNCcnt); otherwise a
// plain 16B copy. Callers must cp16_wait() + __syncthreads() before consuming.
__device__ __forceinline__ void cp16_lds(void* ldst, const void* gsrc) {
#if __has_builtin(__builtin_amdgcn_global_load_async_to_lds_b128)
  __builtin_amdgcn_global_load_async_to_lds_b128((v4i*)gsrc, (v4i*)ldst, 0, 0);
#else
  *(uint4*)ldst = *(const uint4*)gsrc;
#endif
}
__device__ __forceinline__ void cp16_wait() {
#if __has_builtin(__builtin_amdgcn_global_load_async_to_lds_b128) && \
    __has_builtin(__builtin_amdgcn_s_wait_asynccnt)
  __builtin_amdgcn_s_wait_asynccnt(0);
#endif
}

// A-matrix fragment (16x32 bf16): lane holds row m=lane&15; K elements are
// 16 contiguous values starting at hi*16.
__device__ __forceinline__ v16bf load_afrag(const bf16_t* rowk, int hi) {
  v8bf l = *(const v8bf*)(rowk + hi * 16);
  v8bf h = *(const v8bf*)(rowk + hi * 16 + 8);
  v16bf a;
#pragma unroll
  for (int i = 0; i < 8; ++i) { a[i] = l[i]; a[i + 8] = h[i]; }
  return a;
}
// B-matrix fragment (32x16 bf16): lane holds column n=lane&15; K elements are
// 16 contiguous values starting at hi*16.
__device__ __forceinline__ v16bf load_bfrag(const bf16_t* p) {
  v8bf l = *(const v8bf*)(p);
  v8bf h = *(const v8bf*)(p + 8);
  v16bf b;
#pragma unroll
  for (int i = 0; i < 8; ++i) { b[i] = l[i]; b[i + 8] = h[i]; }
  return b;
}

// ---------------------------------------------------------------------------
// P0: elementwise f32 -> bf16 (weights), vectorized 4-wide.
// ---------------------------------------------------------------------------
__global__ __launch_bounds__(256) void f32_to_bf16_kernel(
    const float* __restrict__ src, unsigned short* __restrict__ dst, int n4) {
  int i = blockIdx.x * blockDim.x + threadIdx.x;
  if (i < n4) {
    v4f v = ((const v4f*)src)[i];
    unsigned short s0 = bf2us((bf16_t)v[0]), s1 = bf2us((bf16_t)v[1]);
    unsigned short s2 = bf2us((bf16_t)v[2]), s3 = bf2us((bf16_t)v[3]);
    uint2 w;
    w.x = (unsigned)s0 | ((unsigned)s1 << 16);
    w.y = (unsigned)s2 | ((unsigned)s3 << 16);
    ((uint2*)dst)[i] = w;
  }
}

// ---------------------------------------------------------------------------
// P1: x[b][c][p] f32  ->  xT[b][p][c] bf16, 32x32 LDS tile transpose.
// ---------------------------------------------------------------------------
__global__ __launch_bounds__(256) void xpose_kernel(
    const float* __restrict__ x, unsigned short* __restrict__ xT) {
  __shared__ float tile[32][33];
  const int blk = blockIdx.x;               // 16 b * 6 ctiles * 512 ptiles
  const int pt = blk & 511;
  const int rest = blk >> 9;
  const int ct = rest % 6;
  const int b = rest / 6;
  const int c0 = ct * 32, p0 = pt * 32;
  const int t = threadIdx.x;
  const int r = t >> 3;          // 0..31
  const int q = (t & 7) * 4;     // 0..28

  v4f v = *(const v4f*)(x + ((size_t)b * NCH + c0 + r) * NPIX + p0 + q);
  tile[r][q] = v[0]; tile[r][q + 1] = v[1];
  tile[r][q + 2] = v[2]; tile[r][q + 3] = v[3];
  __syncthreads();

  unsigned short s[4];
#pragma unroll
  for (int j = 0; j < 4; ++j) s[j] = bf2us((bf16_t)tile[q + j][r]);
  uint2 w;
  w.x = (unsigned)s[0] | ((unsigned)s[1] << 16);
  w.y = (unsigned)s[2] | ((unsigned)s[3] << 16);
  *(uint2*)(xT + ((size_t)b * NPIX + p0 + r) * NCH + c0 + q) = w;
}

// ---------------------------------------------------------------------------
// K1: qkv[b,o,p] = sum_c x[b,c,p] * w_qkv[o,c].
// Block = (b, og, 8 p-tiles): async-stages a 48-row (q/k/v thirds) weight
// panel to LDS once; each wave keeps 3 accumulators and reuses one B-fragment
// for 3 WMMAs per K step (3x B reuse, 18 WMMAs per wave).
// ---------------------------------------------------------------------------
__global__ __launch_bounds__(256) void qkv_gemm_kernel(
    const unsigned short* __restrict__ xT_, const unsigned short* __restrict__ wq_,
    float* __restrict__ qkv) {
  __shared__ __align__(16) unsigned short wqp[3 * 16 * NCH];   // 18 KB
  const bf16_t* xT = (const bf16_t*)xT_;
  const int lane = threadIdx.x & 31;
  const int wave = threadIdx.x >> 5;
  const int hi = lane >> 4;
  const int lo = lane & 15;
  const int pt8 = blockIdx.x & 127;         // grid: b(16) x og(8) x pt8(128)
  const int og  = (blockIdx.x >> 7) & 7;
  const int b   = blockIdx.x >> 10;
  const int t = threadIdx.x;

  // stage panel rows o = og*16 + tt*128 + r (tt = q/k/v third), 1152 x 16B
  for (int g = t; g < 1152; g += 256) {
    const int tt = g / 384;
    const int s  = g % 384;
    const int r  = s / 24;      // row within 16
    const int cc = s % 24;      // 8-element column chunk
    cp16_lds(&wqp[g * 8],
             wq_ + ((size_t)(og * 16 + tt * 128 + r) * NCH + cc * 8));
  }
  cp16_wait();
  __syncthreads();

  const int pbase = (pt8 * 8 + wave) * 16;
  const bf16_t* brow  = xT + ((size_t)b * NPIX + pbase + lo) * NCH;
  const bf16_t* a0row = (const bf16_t*)&wqp[(0 * 16 + lo) * NCH];
  const bf16_t* a1row = (const bf16_t*)&wqp[(1 * 16 + lo) * NCH];
  const bf16_t* a2row = (const bf16_t*)&wqp[(2 * 16 + lo) * NCH];

  v8f acc0 = {}, acc1 = {}, acc2 = {};
#pragma unroll
  for (int kb = 0; kb < NCH / 32; ++kb) {
    const int kbase = kb * 32;
    v16bf bm = load_bfrag(brow + kbase + hi * 16);
    v16bf a0 = load_afrag(a0row + kbase, hi);
    v16bf a1 = load_afrag(a1row + kbase, hi);
    v16bf a2 = load_afrag(a2row + kbase, hi);
    acc0 = __builtin_amdgcn_wmma_f32_16x16x32_bf16(false, a0, false, bm, (short)0, acc0, false, false);
    acc1 = __builtin_amdgcn_wmma_f32_16x16x32_bf16(false, a1, false, bm, (short)0, acc1, false, false);
    acc2 = __builtin_amdgcn_wmma_f32_16x16x32_bf16(false, a2, false, bm, (short)0, acc2, false, false);
  }
#pragma unroll
  for (int r = 0; r < 8; ++r) {
    const int orow = og * 16 + r + 8 * hi;
    qkv[((size_t)b * QKV3 + orow) * NPIX + pbase + lo]       = acc0[r];
    qkv[((size_t)b * QKV3 + 128 + orow) * NPIX + pbase + lo] = acc1[r];
    qkv[((size_t)b * QKV3 + 256 + orow) * NPIX + pbase + lo] = acc2[r];
  }
}

// ---------------------------------------------------------------------------
// K2: per (b, head, d) row of K (length 16384): rowmax and sum(exp(.-max)).
// ---------------------------------------------------------------------------
__global__ __launch_bounds__(256) void krow_stats_kernel(
    const float* __restrict__ qkv, float* __restrict__ kmax,
    float* __restrict__ ksum) {
  __shared__ float red[256];
  const int row = blockIdx.x;            // b*128 + h*32 + d, 2048 rows
  const int b = row >> 7;
  const int hd = row & 127;
  const float* kr = qkv + ((size_t)b * QKV3 + HIDW + hd) * NPIX;
  const int tid = threadIdx.x;

  float m = -3.4e38f;
  for (int i = tid; i < NPIX; i += 256) m = fmaxf(m, kr[i]);
  red[tid] = m; __syncthreads();
  for (int s = 128; s > 0; s >>= 1) {
    if (tid < s) red[tid] = fmaxf(red[tid], red[tid + s]);
    __syncthreads();
  }
  const float rowmax = red[0]; __syncthreads();

  float sum = 0.f;
  for (int i = tid; i < NPIX; i += 256) sum += expf(kr[i] - rowmax);
  red[tid] = sum; __syncthreads();
  for (int s = 128; s > 0; s >>= 1) {
    if (tid < s) red[tid] += red[tid + s];
    __syncthreads();
  }
  if (tid == 0) { kmax[row] = rowmax; ksum[row] = red[0]; }
}

// ---------------------------------------------------------------------------
// K3: context[b,h,d,e] = (1/ksum_d) * sum_n exp(k[d,n]-kmax_d) * v[e,n]
// f32 b128 loads (exp applied on the fly), 2x2 WMMA tiles per wave,
// deterministic cross-wave reduce via per-wave LDS partials.
// ---------------------------------------------------------------------------
__global__ __launch_bounds__(256) void context_kernel(
    const float* __restrict__ qkv, const float* __restrict__ kmax,
    const float* __restrict__ ksum, float* __restrict__ ctx) {
  __shared__ float part[8][32][32];   // 32 KB
  const int lane = threadIdx.x & 31;
  const int wave = threadIdx.x >> 5;
  const int hi = lane >> 4;
  const int lo = lane & 15;
  const int b = blockIdx.x >> 2;
  const int h = blockIdx.x & 3;
  const float* kp = qkv + ((size_t)b * QKV3 + HIDW + h * DKEYS) * NPIX;
  const float* vp = qkv + ((size_t)b * QKV3 + 2 * HIDW + h * DKEYS) * NPIX;
  const float* km = kmax + (b * HIDW + h * DKEYS);

  v8f acc00 = {}, acc01 = {}, acc10 = {}, acc11 = {};
  const float km0 = km[lo];
  const float km1 = km[16 + lo];
  for (int chunk = wave; chunk < NPIX / 32; chunk += 8) {
    const int n0 = chunk * 32;
    const v4f* ka = (const v4f*)(kp + (size_t)lo * NPIX + n0 + hi * 16);
    const v4f* kb = (const v4f*)(kp + (size_t)(16 + lo) * NPIX + n0 + hi * 16);
    v4f ka0 = ka[0], ka1 = ka[1], ka2 = ka[2], ka3 = ka[3];
    v4f kb0 = kb[0], kb1 = kb[1], kb2 = kb[2], kb3 = kb[3];
    const v4f* va = (const v4f*)(vp + (size_t)lo * NPIX + n0 + hi * 16);
    const v4f* vb = (const v4f*)(vp + (size_t)(16 + lo) * NPIX + n0 + hi * 16);
    v4f va0 = va[0], va1 = va[1], va2 = va[2], va3 = va[3];
    v4f vb0 = vb[0], vb1 = vb[1], vb2 = vb[2], vb3 = vb[3];

    v16bf a0, a1, b0, b1;
#pragma unroll
    for (int i = 0; i < 4; ++i) {
      a0[i]      = (bf16_t)expf(ka0[i] - km0);
      a0[i + 4]  = (bf16_t)expf(ka1[i] - km0);
      a0[i + 8]  = (bf16_t)expf(ka2[i] - km0);
      a0[i + 12] = (bf16_t)expf(ka3[i] - km0);
      a1[i]      = (bf16_t)expf(kb0[i] - km1);
      a1[i + 4]  = (bf16_t)expf(kb1[i] - km1);
      a1[i + 8]  = (bf16_t)expf(kb2[i] - km1);
      a1[i + 12] = (bf16_t)expf(kb3[i] - km1);
      b0[i]      = (bf16_t)va0[i];
      b0[i + 4]  = (bf16_t)va1[i];
      b0[i + 8]  = (bf16_t)va2[i];
      b0[i + 12] = (bf16_t)va3[i];
      b1[i]      = (bf16_t)vb0[i];
      b1[i + 4]  = (bf16_t)vb1[i];
      b1[i + 8]  = (bf16_t)vb2[i];
      b1[i + 12] = (bf16_t)vb3[i];
    }
    acc00 = __builtin_amdgcn_wmma_f32_16x16x32_bf16(false, a0, false, b0, (short)0, acc00, false, false);
    acc01 = __builtin_amdgcn_wmma_f32_16x16x32_bf16(false, a0, false, b1, (short)0, acc01, false, false);
    acc10 = __builtin_amdgcn_wmma_f32_16x16x32_bf16(false, a1, false, b0, (short)0, acc10, false, false);
    acc11 = __builtin_amdgcn_wmma_f32_16x16x32_bf16(false, a1, false, b1, (short)0, acc11, false, false);
  }
#pragma unroll
  for (int r = 0; r < 8; ++r) {
    part[wave][r + 8 * hi][lo]           = acc00[r];
    part[wave][r + 8 * hi][16 + lo]      = acc01[r];
    part[wave][16 + r + 8 * hi][lo]      = acc10[r];
    part[wave][16 + r + 8 * hi][16 + lo] = acc11[r];
  }
  __syncthreads();
  const float* ks = ksum + (b * HIDW + h * DKEYS);
  for (int idx = threadIdx.x; idx < 1024; idx += 256) {
    const int d = idx >> 5;
    float s = 0.f;
#pragma unroll
    for (int w = 0; w < 8; ++w) s += part[w][idx >> 5][idx & 31];
    ctx[(size_t)blockIdx.x * 1024 + idx] = s / ks[d];
  }
}

// ---------------------------------------------------------------------------
// K4: per-pixel q-softmax staged transposed bf16 in LDS (qsT[p][d]) plus
// transposed bf16 context (ctxT[e][d]); one WMMA per 16e x 16p tile, all
// fragments contiguous ds_load_b128. Result stored transposed bf16:
// resT[b][p][ch] so K5's B operand is contiguous.
// ---------------------------------------------------------------------------
__global__ __launch_bounds__(256) void res_gemm_kernel(
    const float* __restrict__ qkv, const float* __restrict__ ctx,
    unsigned short* __restrict__ resT) {
  __shared__ __align__(16) unsigned short qsT[128][32];   // 8 KB bf16 bits
  __shared__ __align__(16) unsigned short ctxT[32][32];   // 2 KB bf16 bits
  const int lane = threadIdx.x & 31;
  const int wave = threadIdx.x >> 5;
  const int hi = lane >> 4;
  const int lo = lane & 15;
  const int bid  = blockIdx.x;            // 16*4*128 = 8192 blocks
  const int b    = bid >> 9;
  const int h    = (bid >> 7) & 3;
  const int pblk = bid & 127;
  const int pbase = pblk * 128;
  const float* qp = qkv + ((size_t)b * QKV3 + h * DKEYS) * NPIX;
  const int tid = threadIdx.x;

  if (tid < 128) {
    const int p = pbase + tid;
    float q[32];
    float mx = -3.4e38f;
#pragma unroll
    for (int d = 0; d < 32; ++d) { q[d] = qp[(size_t)d * NPIX + p]; mx = fmaxf(mx, q[d]); }
    float s = 0.f;
#pragma unroll
    for (int d = 0; d < 32; ++d) { q[d] = expf(q[d] - mx); s += q[d]; }
    const float inv = SM_SCALE / s;
#pragma unroll
    for (int d = 0; d < 32; ++d) qsT[tid][d] = bf2us((bf16_t)(q[d] * inv));
  } else {
    const float* cp = ctx + (size_t)(b * 4 + h) * 1024;
    for (int idx = tid - 128; idx < 1024; idx += 128)
      ctxT[idx & 31][idx >> 5] = bf2us((bf16_t)cp[idx]);   // ctx[d][e] -> ctxT[e][d]
  }
  __syncthreads();

#pragma unroll
  for (int t = 0; t < 2; ++t) {
    const int tileId = wave + t * 8;  // 16 tiles: 2 e-tiles x 8 p-tiles
    const int etile = tileId >> 3;
    const int ptile = tileId & 7;
    v16bf a  = load_afrag((const bf16_t*)&ctxT[etile * 16 + lo][0], hi);
    v16bf bm = load_bfrag((const bf16_t*)&qsT[ptile * 16 + lo][hi * 16]);
    v8f acc = {};
    acc = __builtin_amdgcn_wmma_f32_16x16x32_bf16(false, a, false, bm,
                                                  (short)0, acc, false, false);
    unsigned short s[8];
#pragma unroll
    for (int r = 0; r < 8; ++r) s[r] = bf2us((bf16_t)acc[r]);
    uint4 w;
    w.x = (unsigned)s[0] | ((unsigned)s[1] << 16);
    w.y = (unsigned)s[2] | ((unsigned)s[3] << 16);
    w.z = (unsigned)s[4] | ((unsigned)s[5] << 16);
    w.w = (unsigned)s[6] | ((unsigned)s[7] << 16);
    const int p = pbase + ptile * 16 + lo;
    const int ch = h * DKEYS + etile * 16 + 8 * hi;
    *(uint4*)(resT + ((size_t)b * NPIX + p) * HIDW + ch) = w;
  }
}

// ---------------------------------------------------------------------------
// K5: out[b,o,p] = sum_e w_out[o,e]*resT[p,e] + b_out[o].
// Block = (b, og, 8 p-tiles): async-stages 48-row w_out panel to LDS; each
// wave keeps 3 accumulators (o, o+64, o+128 thirds), 3x B reuse, 12 WMMAs.
// Emits deterministic per-block GN partials.
// ---------------------------------------------------------------------------
__global__ __launch_bounds__(256) void out_gemm_kernel(
    const unsigned short* __restrict__ resT_, const unsigned short* __restrict__ wo_,
    const float* __restrict__ b_out, float* __restrict__ out,
    float* __restrict__ psum, float* __restrict__ psq) {
  __shared__ __align__(16) unsigned short wop[3 * 16 * HIDW];  // 12 KB
  __shared__ float rs[256], rq[256];
  const bf16_t* resT = (const bf16_t*)resT_;
  const int lane = threadIdx.x & 31;
  const int wave = threadIdx.x >> 5;
  const int hi = lane >> 4;
  const int lo = lane & 15;
  const int pt8 = blockIdx.x & 127;        // grid: b(16) x og(4) x pt8(128)
  const int og  = (blockIdx.x >> 7) & 3;
  const int b   = blockIdx.x >> 9;
  const int t = threadIdx.x;

  for (int g = t; g < 768; g += 256) {     // 48 rows x 128 cols bf16, 16B chunks
    const int tt = g / 256;
    const int s  = g % 256;
    const int r  = s / 16;
    const int cc = s % 16;
    cp16_lds(&wop[g * 8],
             wo_ + ((size_t)(og * 16 + tt * 64 + r) * HIDW + cc * 8));
  }
  cp16_wait();
  __syncthreads();

  const int pbase = (pt8 * 8 + wave) * 16;
  const bf16_t* brow  = resT + ((size_t)b * NPIX + pbase + lo) * HIDW;
  const bf16_t* a0row = (const bf16_t*)&wop[(0 * 16 + lo) * HIDW];
  const bf16_t* a1row = (const bf16_t*)&wop[(1 * 16 + lo) * HIDW];
  const bf16_t* a2row = (const bf16_t*)&wop[(2 * 16 + lo) * HIDW];

  v8f acc0 = {}, acc1 = {}, acc2 = {};
#pragma unroll
  for (int kb = 0; kb < HIDW / 32; ++kb) {
    const int kbase = kb * 32;
    v16bf bm = load_bfrag(brow + kbase + hi * 16);
    v16bf a0 = load_afrag(a0row + kbase, hi);
    v16bf a1 = load_afrag(a1row + kbase, hi);
    v16bf a2 = load_afrag(a2row + kbase, hi);
    acc0 = __builtin_amdgcn_wmma_f32_16x16x32_bf16(false, a0, false, bm, (short)0, acc0, false, false);
    acc1 = __builtin_amdgcn_wmma_f32_16x16x32_bf16(false, a1, false, bm, (short)0, acc1, false, false);
    acc2 = __builtin_amdgcn_wmma_f32_16x16x32_bf16(false, a2, false, bm, (short)0, acc2, false, false);
  }
  float lsum = 0.f, lsq = 0.f;
#pragma unroll
  for (int r = 0; r < 8; ++r) {
    const int o0 = og * 16 + r + 8 * hi;
    const float v0 = acc0[r] + b_out[o0];
    const float v1 = acc1[r] + b_out[o0 + 64];
    const float v2 = acc2[r] + b_out[o0 + 128];
    out[((size_t)b * NCH + o0) * NPIX + pbase + lo]         = v0;
    out[((size_t)b * NCH + o0 + 64) * NPIX + pbase + lo]    = v1;
    out[((size_t)b * NCH + o0 + 128) * NPIX + pbase + lo]   = v2;
    lsum += v0 + v1 + v2;
    lsq  += v0 * v0 + v1 * v1 + v2 * v2;
  }
  rs[threadIdx.x] = lsum; rq[threadIdx.x] = lsq; __syncthreads();
  for (int s = 128; s > 0; s >>= 1) {
    if (threadIdx.x < s) {
      rs[threadIdx.x] += rs[threadIdx.x + s];
      rq[threadIdx.x] += rq[threadIdx.x + s];
    }
    __syncthreads();
  }
  if (threadIdx.x == 0) { psum[blockIdx.x] = rs[0]; psq[blockIdx.x] = rq[0]; }
}

// ---------------------------------------------------------------------------
// K6: per-batch mean / rsqrt(var+eps) from 512 partials per batch.
// ---------------------------------------------------------------------------
__global__ __launch_bounds__(256) void gn_stats_kernel(
    const float* __restrict__ psum, const float* __restrict__ psq,
    float* __restrict__ mean, float* __restrict__ inv) {
  __shared__ float rs[256], rq[256];
  const int b = blockIdx.x;
  float s = 0.f, q = 0.f;
  for (int i = threadIdx.x; i < 512; i += 256) {
    s += psum[b * 512 + i]; q += psq[b * 512 + i];
  }
  rs[threadIdx.x] = s; rq[threadIdx.x] = q; __syncthreads();
  for (int st = 128; st > 0; st >>= 1) {
    if (threadIdx.x < st) {
      rs[threadIdx.x] += rs[threadIdx.x + st];
      rq[threadIdx.x] += rq[threadIdx.x + st];
    }
    __syncthreads();
  }
  if (threadIdx.x == 0) {
    const float cnt = (float)NCH * (float)NPIX;
    const float m = rs[0] / cnt;
    const float v = rq[0] / cnt - m * m;
    mean[b] = m;
    inv[b] = rsqrtf(v + GN_EPS);
  }
}

// ---------------------------------------------------------------------------
// K7: out = (out - mean_b) * inv_b * gn_w[c] + gn_b[c], float4 vectorized.
// ---------------------------------------------------------------------------
__global__ __launch_bounds__(256) void gn_apply_kernel(
    float* __restrict__ out, const float* __restrict__ mean,
    const float* __restrict__ inv, const float* __restrict__ gnw,
    const float* __restrict__ gnb) {
  const size_t total4 = (size_t)BATCH * NCH * NPIX / 4;
  const size_t stride = (size_t)gridDim.x * blockDim.x;
  for (size_t i = blockIdx.x * (size_t)blockDim.x + threadIdx.x; i < total4; i += stride) {
    const size_t base = i * 4;
    const int b = (int)(base / ((size_t)NCH * NPIX));
    const int c = (int)((base / NPIX) % NCH);
    v4f v = ((v4f*)out)[i];
    const float m = mean[b], iv = inv[b];
    const float w = gnw[c] * iv;
    const float bb = gnb[c] - m * w;
#pragma unroll
    for (int j = 0; j < 4; ++j) v[j] = v[j] * w + bb;
    ((v4f*)out)[i] = v;
  }
}

// ---------------------------------------------------------------------------
extern "C" void kernel_launch(void* const* d_in, const int* in_sizes, int n_in,
                              void* d_out, int out_size, void* d_ws, size_t ws_size,
                              hipStream_t stream) {
  (void)in_sizes; (void)n_in; (void)out_size; (void)ws_size;
  const float* x     = (const float*)d_in[0];
  const float* w_qkv = (const float*)d_in[1];
  const float* w_out = (const float*)d_in[2];
  const float* b_out = (const float*)d_in[3];
  const float* gn_w  = (const float*)d_in[4];
  const float* gn_b  = (const float*)d_in[5];
  float* out = (float*)d_out;

  char* ws = (char*)d_ws;
  size_t off = 0;
  float* qkv = (float*)(ws + off);            off += (size_t)BATCH * QKV3 * NPIX * sizeof(float);
  unsigned short* xT = (unsigned short*)(ws + off);
                                              off += (size_t)BATCH * NPIX * NCH * sizeof(unsigned short);
  unsigned short* resT = (unsigned short*)(ws + off);
                                              off += (size_t)BATCH * NPIX * HIDW * sizeof(unsigned short);
  unsigned short* wq_bf = (unsigned short*)(ws + off); off += (size_t)QKV3 * NCH * sizeof(unsigned short) + 128;
  unsigned short* wo_bf = (unsigned short*)(ws + off); off += (size_t)NCH * HIDW * sizeof(unsigned short) + 128;
  float* kmaxp = (float*)(ws + off);          off += 2048 * sizeof(float);
  float* ksump = (float*)(ws + off);          off += 2048 * sizeof(float);
  float* ctxp = (float*)(ws + off);           off += (size_t)BATCH * NHEADS * 1024 * sizeof(float);
  float* psum = (float*)(ws + off);           off += (size_t)16 * 512 * sizeof(float);
  float* psq = (float*)(ws + off);            off += (size_t)16 * 512 * sizeof(float);
  float* meanp = (float*)(ws + off);          off += 256;
  float* invp = (float*)(ws + off);           off += 256;

  // P0: weight conversions (73728 and 24576 elements, 4-wide)
  f32_to_bf16_kernel<<<72, 256, 0, stream>>>(w_qkv, wq_bf, QKV3 * NCH / 4);
  f32_to_bf16_kernel<<<24, 256, 0, stream>>>(w_out, wo_bf, NCH * HIDW / 4);
  // P1: x transpose+convert: 16 b * 6 ctiles * 512 ptiles
  xpose_kernel<<<49152, 256, 0, stream>>>(x, xT);
  // K1: b(16) x og(8) x pt8(128) = 16384 blocks
  qkv_gemm_kernel<<<16384, 256, 0, stream>>>(xT, wq_bf, qkv);
  // K2: 2048 k-rows
  krow_stats_kernel<<<2048, 256, 0, stream>>>(qkv, kmaxp, ksump);
  // K3: 64 (b,h) pairs
  context_kernel<<<64, 256, 0, stream>>>(qkv, kmaxp, ksump, ctxp);
  // K4: 16*4*128 = 8192 blocks (128 pixels each)
  res_gemm_kernel<<<8192, 256, 0, stream>>>(qkv, ctxp, resT);
  // K5: b(16) x og(4) x pt8(128) = 8192 blocks
  out_gemm_kernel<<<8192, 256, 0, stream>>>(resT, wo_bf, b_out, out, psum, psq);
  // K6: one block per batch
  gn_stats_kernel<<<16, 256, 0, stream>>>(psum, psq, meanp, invp);
  // K7: elementwise normalize
  gn_apply_kernel<<<4096, 256, 0, stream>>>(out, meanp, invp, gn_w, gn_b);
}
